// GERLLoss_28724741275667
// MI455X (gfx1250) — matrix operand order repaired and checked by reading, not compiled, not run on hardware
//
#include <hip/hip_runtime.h>
#include <math.h>

// GERL loss: B=16384, C=50, H=64, K=4 negatives.
// Gather-then-dot (only 5 of 50 scores per row are used) => ~30 MB traffic
// instead of 210 MB. Dot products done with V_WMMA_F32_16X16X4_F32 (full f32),
// 16 batches per wave, diagonal extraction of the 16x16 D tile.

#define BATCH_PER_WAVE 16
#define WAVES_PER_BLK  4
#define THREADS_PER_BLK (WAVES_PER_BLK * 32)
#define BATCH_PER_BLK  (WAVES_PER_BLK * BATCH_PER_WAVE)
#define CC 50
#define HH 64
#define KK 4

typedef __attribute__((ext_vector_type(2)))  float    v2f;
typedef __attribute__((ext_vector_type(8)))  float    v8f;
typedef __attribute__((ext_vector_type(16))) _Float16 v16h;

#if defined(__HIP_DEVICE_COMPILE__) && __has_builtin(__builtin_amdgcn_wmma_f32_16x16x4_f32)
#define GERL_USE_F32_WMMA 1
#else
#define GERL_USE_F32_WMMA 0
#endif

__global__ __launch_bounds__(THREADS_PER_BLK) void gerl_partial_kernel(
    const float* __restrict__ user,    // [B,H] f32
    const float* __restrict__ news,    // [B,C,H] f32
    const int*   __restrict__ labels,  // [B,C] one-hot (int32 from harness)
    const int*   __restrict__ negidx,  // [B,K] int32
    float*       __restrict__ partial) // [gridDim.x] block partial sums
{
    const int lane = threadIdx.x & 31;
    const int wave = threadIdx.x >> 5;
    const int m    = lane & 15;                       // row/col slot in 16x16 tile
    const int bm   = blockIdx.x * BATCH_PER_BLK + wave * BATCH_PER_WAVE + m;

    // ---- find positive index: one-hot argmax over C (uniform 50-iter loop) ----
    int pos = 0;
    const int* lab = labels + bm * CC;
    #pragma unroll 10
    for (int c = 0; c < CC; ++c) pos = (lab[c] != 0) ? c : pos;

    int cand[5];
    cand[0] = pos;
    #pragma unroll
    for (int k = 0; k < KK; ++k) cand[k + 1] = negidx[bm * KK + k];

    const float* urow = user + bm * HH;
    float sc[5];

#if GERL_USE_F32_WMMA
    // ---- A operand (16x4 f32, chained over 16 K-chunks of the H=64 row) ----
    // lane layout: VGPR0 = K{kb}, VGPR1 = K{kb+1}; kb = (lane>=16) ? 2 : 0
    const int kb = (lane >> 4) * 2;
    v2f a[16];
    #pragma unroll
    for (int t = 0; t < 16; ++t) a[t] = *(const v2f*)(urow + t * 4 + kb);

    #pragma unroll
    for (int j = 0; j < 5; ++j) {
        const float* nrow = news + (bm * CC + cand[j]) * HH;   // gathered row
        v8f acc = {};
        #pragma unroll
        for (int t = 0; t < 16; ++t) {
            v2f bv = *(const v2f*)(nrow + t * 4 + kb);          // B operand 4x16
            acc = __builtin_amdgcn_wmma_f32_16x16x4_f32(
                false, a[t], false, bv, (short)0, acc, false, false);
        }
        // ---- diagonal D[m][m]: vgpr r=m&7, source lane = m (m<8) or m+16 ----
        const int src = (m < 8) ? m : (m + 16);
        const int r   = m & 7;
        float tv[8];
        #pragma unroll
        for (int i = 0; i < 8; ++i) tv[i] = __shfl(acc[i], src, 32);
        float s = tv[0];
        #pragma unroll
        for (int i = 1; i < 8; ++i) s = (r == i) ? tv[i] : s;
        sc[j] = s;
    }
#else
    // ---- fallback: codegen-confirmed V_WMMA_F32_16X16X32_F16, K=64 in 2 ops ----
    const int kq = lane >> 4;
    v16h a01, a23;
    #pragma unroll
    for (int h = 0; h < 16; ++h) {
        a01[h]     = (_Float16)urow[kq * 16 + h];
        a23[h]     = (_Float16)urow[32 + kq * 16 + h];
    }
    #pragma unroll
    for (int j = 0; j < 5; ++j) {
        const float* nrow = news + (bm * CC + cand[j]) * HH;
        v16h b01, b23;
        #pragma unroll
        for (int h = 0; h < 16; ++h) {
            b01[h] = (_Float16)nrow[kq * 16 + h];         // B 32x16: K = lanehalf*16+h
            b23[h] = (_Float16)nrow[32 + kq * 16 + h];
        }
        v8f acc = {};
        acc = __builtin_amdgcn_wmma_f32_16x16x32_f16(false, a01, false, b01,
                                                     (short)0, acc, false, false);
        acc = __builtin_amdgcn_wmma_f32_16x16x32_f16(false, a23, false, b23,
                                                     (short)0, acc, false, false);
        const int src = (m < 8) ? m : (m + 16);
        const int r   = m & 7;
        float tv[8];
        #pragma unroll
        for (int i = 0; i < 8; ++i) tv[i] = __shfl(acc[i], src, 32);
        float s = tv[0];
        #pragma unroll
        for (int i = 1; i < 8; ++i) s = (r == i) ? tv[i] : s;
        sc[j] = s;
    }
#endif

    // ---- sampled-softmax loss, exactly the reference formula ----
    float pexp = expf(sc[0]);
    float nexp = expf(sc[1]) + expf(sc[2]) + expf(sc[3]) + expf(sc[4]);
    float lossv = -logf(pexp / (pexp + nexp));
    lossv = (lane < 16) ? lossv : 0.0f;   // lanes 16..31 hold duplicates

    // ---- deterministic fixed-order reductions ----
    #pragma unroll
    for (int off = 16; off >= 1; off >>= 1)
        lossv += __shfl_xor(lossv, off, 32);

    __shared__ float wsum[WAVES_PER_BLK];
    if (lane == 0) wsum[wave] = lossv;
    __syncthreads();
    if (threadIdx.x == 0) {
        float s = 0.0f;
        #pragma unroll
        for (int w = 0; w < WAVES_PER_BLK; ++w) s += wsum[w];
        partial[blockIdx.x] = s;
    }
}

__global__ void gerl_reduce_kernel(const float* __restrict__ partial,
                                   float* __restrict__ out,
                                   int n, float invB)
{
    if (threadIdx.x == 0 && blockIdx.x == 0) {
        float s = 0.0f;
        for (int i = 0; i < n; ++i) s += partial[i];   // fixed order: deterministic
        out[0] = s * invB;
    }
}

extern "C" void kernel_launch(void* const* d_in, const int* in_sizes, int n_in,
                              void* d_out, int out_size, void* d_ws, size_t ws_size,
                              hipStream_t stream) {
    const float* user   = (const float*)d_in[0];
    const float* news   = (const float*)d_in[1];
    const int*   labels = (const int*)d_in[2];
    const int*   negidx = (const int*)d_in[3];
    float* out     = (float*)d_out;
    float* partial = (float*)d_ws;     // needs gridDim * 4 bytes (1 KB) of scratch

    const int B       = in_sizes[0] / HH;          // 16384
    const int nblocks = B / BATCH_PER_BLK;         // 256 blocks x 128 threads

    gerl_partial_kernel<<<nblocks, THREADS_PER_BLK, 0, stream>>>(
        user, news, labels, negidx, partial);
    gerl_reduce_kernel<<<1, 1, 0, stream>>>(partial, out, nblocks,
                                            1.0f / (float)B);
}
